// GeneInteractionAttention_76252849373275
// MI455X (gfx1250) — matrix operand order repaired and checked by reading, not compiled, not run on hardware
//
#include <hip/hip_runtime.h>
#include <hip/hip_bf16.h>
#include <math.h>

// Vector types for WMMA operands.
typedef __attribute__((ext_vector_type(16))) _Float16 v16h;
typedef __attribute__((ext_vector_type(8)))  _Float16 v8h;
typedef __attribute__((ext_vector_type(8)))  float    v8f;

#define LPAD 40   // LDS row pitch in f16 (32 data + 8 pad) -> 80B rows, 16B aligned

// CDNA5 async global->LDS copy path (ASYNCcnt), if this toolchain exposes it.
#if defined(__has_builtin)
#if __has_builtin(__builtin_amdgcn_global_load_async_to_lds_b128) && \
    __has_builtin(__builtin_amdgcn_s_wait_asynccnt)
#define USE_ASYNC_LDS 1
#endif
#endif
#ifndef USE_ASYNC_LDS
#define USE_ASYNC_LDS 0
#endif

// Builtin prototype uses GCC-style int __vector(4) pointers:
//   void __builtin_amdgcn_global_load_async_to_lds_b128(v4i AS(1)*, v4i AS(3)*, Ii, Ii)
typedef int b128_t __attribute__((vector_size(16)));
typedef __attribute__((address_space(1))) b128_t ga_b128;   // global 16B chunk
typedef __attribute__((address_space(3))) b128_t ls_b128;   // LDS 16B chunk

// ---------------------------------------------------------------------------
// f32 -> f16 cast (grid-stride)
// ---------------------------------------------------------------------------
__global__ __launch_bounds__(256)
void cast_f32_to_f16(const float* __restrict__ in, _Float16* __restrict__ out, long n) {
  long i = (long)blockIdx.x * blockDim.x + threadIdx.x;
  long stride = (long)gridDim.x * blockDim.x;
  for (; i < n; i += stride) out[i] = (_Float16)in[i];
}

// ---------------------------------------------------------------------------
// Generic WMMA GEMM:  C[M,N] = A[M,K] * B[N,K]^T   (both row-major, K contiguous)
// 128x128 tile per block, 256 threads = 8 waves (2x4), each wave: 4x2 WMMA tiles.
// Double-buffered LDS; async global->LDS copies overlap the WMMA compute.
// modes:
//   0: Ch[z*sC + m*ldc + n] = f16( acc*scale + bias[n] )
//   1: Ch[((m/GS)*N + n)*GS + (m%GS)] = f16( acc + bias[n] )   (grouped transpose, V^T)
//   2: Cf[z*sC + m*ldc + n] = acc*scale
//   3: Cf[m*ldc + n] = Xres[m*ldc+n] + beta*(acc + bias[n])     (final residual)
// ---------------------------------------------------------------------------
__global__ __launch_bounds__(256)
void gemm_wmma(const _Float16* __restrict__ A, const _Float16* __restrict__ B,
               const float* __restrict__ bias, _Float16* __restrict__ Ch,
               float* __restrict__ Cf, const float* __restrict__ Xres,
               const float* __restrict__ betaPtr,
               int N, int K, int lda, int ldb, int ldc,
               long sA, long sB, long sC,
               float scale, int mode, int GS) {
  const int tid   = threadIdx.x;
  const int lane  = tid & 31;
  const int wave  = tid >> 5;
  const int waveM = wave >> 2;            // 0..1  -> 64-row slab
  const int waveN = wave & 3;             // 0..3  -> 32-col slab
  const int tileM = blockIdx.y * 128;
  const int tileN = blockIdx.x * 128;
  const long z    = blockIdx.z;

  A += z * sA;
  B += z * sB;

  __shared__ _Float16 As[2][128 * LPAD];
  __shared__ _Float16 Bs[2][128 * LPAD];

  v8f acc[4][2] = {};

  const int m16 = lane & 15;
  const int hi  = lane >> 4;

  // Staging assignment: each thread moves two 16B chunks of A and two of B.
  const int row0 = tid >> 2;        // 0..63
  const int row1 = row0 + 64;       // 64..127
  const int ch   = tid & 3;         // 16B chunk within 32-wide K slice

  auto issue_copy = [&](int k0, int b) {
    const _Float16* ga0 = A + (long)(tileM + row0) * lda + k0 + ch * 8;
    const _Float16* ga1 = A + (long)(tileM + row1) * lda + k0 + ch * 8;
    const _Float16* gb0 = B + (long)(tileN + row0) * ldb + k0 + ch * 8;
    const _Float16* gb1 = B + (long)(tileN + row1) * ldb + k0 + ch * 8;
    _Float16* la0 = &As[b][row0 * LPAD + ch * 8];
    _Float16* la1 = &As[b][row1 * LPAD + ch * 8];
    _Float16* lb0 = &Bs[b][row0 * LPAD + ch * 8];
    _Float16* lb1 = &Bs[b][row1 * LPAD + ch * 8];
#if USE_ASYNC_LDS
    __builtin_amdgcn_global_load_async_to_lds_b128((ga_b128*)ga0, (ls_b128*)la0, 0, 0);
    __builtin_amdgcn_global_load_async_to_lds_b128((ga_b128*)ga1, (ls_b128*)la1, 0, 0);
    __builtin_amdgcn_global_load_async_to_lds_b128((ga_b128*)gb0, (ls_b128*)lb0, 0, 0);
    __builtin_amdgcn_global_load_async_to_lds_b128((ga_b128*)gb1, (ls_b128*)lb1, 0, 0);
#else
    *(v8h*)la0 = *(const v8h*)ga0;
    *(v8h*)la1 = *(const v8h*)ga1;
    *(v8h*)lb0 = *(const v8h*)gb0;
    *(v8h*)lb1 = *(const v8h*)gb1;
#endif
  };

  int buf = 0;
  issue_copy(0, 0);                 // preload first K slice into buffer 0

  for (int k0 = 0; k0 < K; k0 += 32) {
#if USE_ASYNC_LDS
    __builtin_amdgcn_s_wait_asynccnt(0);   // my async fills of As/Bs[buf] landed
#endif
    __syncthreads();                       // everyone's fills landed; prev reads done

    if (k0 + 32 < K) issue_copy(k0 + 32, buf ^ 1);  // overlaps with compute below

    const _Float16* __restrict__ Ab = As[buf];
    const _Float16* __restrict__ Bb = Bs[buf];

    // A fragments: 16x32 f16, lane half 'hi' holds K = hi*8+0..7 then 16+hi*8+0..7
    v16h af[4];
#pragma unroll
    for (int i = 0; i < 4; ++i) {
      int r = waveM * 64 + i * 16 + m16;
      v8h lo  = *(const v8h*)(&Ab[r * LPAD + hi * 8]);
      v8h hi8 = *(const v8h*)(&Ab[r * LPAD + 16 + hi * 8]);
      af[i] = __builtin_shufflevector(lo, hi8, 0, 1, 2, 3, 4, 5, 6, 7,
                                      8, 9, 10, 11, 12, 13, 14, 15);
    }
    // B fragments: 32x16 f16, lane half 'hi' holds K = hi*8+0..7 then 16+hi*8+0..7 (row of B^T tile)
    v16h bf[2];
#pragma unroll
    for (int j = 0; j < 2; ++j) {
      int c = waveN * 32 + j * 16 + m16;
      v8h lo  = *(const v8h*)(&Bb[c * LPAD + hi * 8]);
      v8h hi8 = *(const v8h*)(&Bb[c * LPAD + 16 + hi * 8]);
      bf[j] = __builtin_shufflevector(lo, hi8, 0, 1, 2, 3, 4, 5, 6, 7,
                                      8, 9, 10, 11, 12, 13, 14, 15);
    }
#pragma unroll
    for (int i = 0; i < 4; ++i)
#pragma unroll
      for (int j = 0; j < 2; ++j)
        acc[i][j] = __builtin_amdgcn_wmma_f32_16x16x32_f16(
            false, af[i], false, bf[j], (short)0, acc[i][j], false, false);

    buf ^= 1;
  }

  const float beta = (mode == 3) ? betaPtr[0] : 0.0f;

  // C/D layout: element v of v8f -> row M = v + 8*hi, col N = lane&15
#pragma unroll
  for (int i = 0; i < 4; ++i) {
#pragma unroll
    for (int j = 0; j < 2; ++j) {
      int gm0 = tileM + waveM * 64 + i * 16 + hi * 8;
      int gn  = tileN + waveN * 32 + j * 16 + m16;
      float bv = bias ? bias[gn] : 0.0f;
#pragma unroll
      for (int v = 0; v < 8; ++v) {
        int gm = gm0 + v;
        float val = acc[i][j][v];
        if (mode == 0) {
          Ch[z * sC + (long)gm * ldc + gn] = (_Float16)(val * scale + bv);
        } else if (mode == 1) {
          long g = gm / GS, rem = gm % GS;
          Ch[(g * (long)N + gn) * GS + rem] = (_Float16)(val + bv);
        } else if (mode == 2) {
          Cf[z * sC + (long)gm * ldc + gn] = val * scale;
        } else {
          long off = (long)gm * ldc + gn;
          Cf[off] = Xres[off] + beta * (val + bv);
        }
      }
    }
  }
}

// ---------------------------------------------------------------------------
// Row softmax over 512 entries with diagonal mask (-1e9), f32 in -> f16 out.
// One 512-thread block per row; rows are grouped [16][512][512].
// ---------------------------------------------------------------------------
__global__ __launch_bounds__(512)
void softmax_mask(const float* __restrict__ S, _Float16* __restrict__ P, float scale) {
  int r  = blockIdx.x;        // 0..8191
  int g  = r >> 9;
  int li = r & 511;
  int t  = threadIdx.x;
  long base = ((long)g * 512 + li) * 512;

  float s = S[base + t] * scale;
  if (t == li) s = -1.0e9f;   // self-mask (same fill as reference)

  __shared__ float red[512];
  red[t] = s;
  __syncthreads();
#pragma unroll
  for (int off = 256; off > 0; off >>= 1) {
    if (t < off) red[t] = fmaxf(red[t], red[t + off]);
    __syncthreads();
  }
  float mx = red[0];
  __syncthreads();

  float e = __expf(s - mx);
  red[t] = e;
  __syncthreads();
#pragma unroll
  for (int off = 256; off > 0; off >>= 1) {
    if (t < off) red[t] += red[t + off];
    __syncthreads();
  }
  float sum = red[0];

  P[base + t] = (_Float16)(e / sum);
}

// ---------------------------------------------------------------------------
// Launch: cast -> QKV GEMMs -> scores GEMM -> softmax -> weights*V -> out GEMM
// ---------------------------------------------------------------------------
extern "C" void kernel_launch(void* const* d_in, const int* in_sizes, int n_in,
                              void* d_out, int out_size, void* d_ws, size_t ws_size,
                              hipStream_t stream) {
  const long Nn = 8192, D = 768, G = 16, GS = 512;

  const float* X    = (const float*)d_in[0];
  // d_in[1] = batch (int64): groups are 16 contiguous blocks of 512 -> implicit
  const float* Wq   = (const float*)d_in[2];
  const float* bq   = (const float*)d_in[3];
  const float* Wk   = (const float*)d_in[4];
  const float* bk   = (const float*)d_in[5];
  const float* Wv   = (const float*)d_in[6];
  const float* bv   = (const float*)d_in[7];
  const float* Wo   = (const float*)d_in[8];
  const float* bo   = (const float*)d_in[9];
  const float* beta = (const float*)d_in[10];
  float* out = (float*)d_out;

  // Workspace layout (all sizes are multiples of 256B): ~88.5 MiB total
  char* w = (char*)d_ws;
  _Float16* Xh  = (_Float16*)w; w += Nn * D * 2;
  _Float16* Wqh = (_Float16*)w; w += D * D * 2;
  _Float16* Wkh = (_Float16*)w; w += D * D * 2;
  _Float16* Wvh = (_Float16*)w; w += D * D * 2;
  _Float16* Woh = (_Float16*)w; w += D * D * 2;
  _Float16* Qh  = (_Float16*)w; w += Nn * D * 2;
  _Float16* Kh  = (_Float16*)w; w += Nn * D * 2;
  _Float16* Vth = (_Float16*)w; w += G * D * GS * 2;   // per group: [D][GS] = V^T
  float*    S   = (float*)w;    w += G * GS * GS * 4;  // raw scores f32
  _Float16* Ph  = (_Float16*)w; w += G * GS * GS * 2;  // softmax weights f16
  _Float16* Dh  = (_Float16*)w; w += Nn * D * 2;       // weights @ V

  dim3 blk(256);

  // Casts
  cast_f32_to_f16<<<2048, blk, 0, stream>>>(X,  Xh,  Nn * D);
  cast_f32_to_f16<<<576,  blk, 0, stream>>>(Wq, Wqh, D * D);
  cast_f32_to_f16<<<576,  blk, 0, stream>>>(Wk, Wkh, D * D);
  cast_f32_to_f16<<<576,  blk, 0, stream>>>(Wv, Wvh, D * D);
  cast_f32_to_f16<<<576,  blk, 0, stream>>>(Wo, Woh, D * D);

  // Q = X Wq^T + bq ; K = X Wk^T + bk  (f16 row-major [N,D])
  dim3 gP(D / 128, Nn / 128, 1);
  gemm_wmma<<<gP, blk, 0, stream>>>(Xh, Wqh, bq, Qh, nullptr, nullptr, nullptr,
                                    (int)D, (int)D, (int)D, (int)D, (int)D,
                                    0, 0, 0, 1.0f, 0, (int)GS);
  gemm_wmma<<<gP, blk, 0, stream>>>(Xh, Wkh, bk, Kh, nullptr, nullptr, nullptr,
                                    (int)D, (int)D, (int)D, (int)D, (int)D,
                                    0, 0, 0, 1.0f, 0, (int)GS);
  // V = X Wv^T + bv, stored grouped-transposed: Vth[g][n][m]
  gemm_wmma<<<gP, blk, 0, stream>>>(Xh, Wvh, bv, Vth, nullptr, nullptr, nullptr,
                                    (int)D, (int)D, (int)D, (int)D, (int)D,
                                    0, 0, 0, 1.0f, 1, (int)GS);

  // scores_g = Q_g K_g^T  (f32, unscaled; scale+mask applied in softmax)
  dim3 gS(GS / 128, GS / 128, G);
  gemm_wmma<<<gS, blk, 0, stream>>>(Qh, Kh, nullptr, nullptr, S, nullptr, nullptr,
                                    (int)GS, (int)D, (int)D, (int)D, (int)GS,
                                    GS * D, GS * D, GS * GS, 1.0f, 2, (int)GS);

  // softmax with diagonal mask
  softmax_mask<<<Nn, 512, 0, stream>>>(S, Ph, 1.0f / sqrtf((float)D));

  // dyn_g = P_g V_g  ->  A = P_g [GS,GS], B = V^T_g rows [D,GS]
  dim3 gD(D / 128, GS / 128, G);
  gemm_wmma<<<gD, blk, 0, stream>>>(Ph, Vth, nullptr, Dh, nullptr, nullptr, nullptr,
                                    (int)D, (int)GS, (int)GS, (int)GS, (int)D,
                                    GS * GS, D * GS, GS * D, 1.0f, 0, (int)GS);

  // out = X + beta * (dyn Wo^T + bo)
  dim3 gO(D / 128, Nn / 128, 1);
  gemm_wmma<<<gO, blk, 0, stream>>>(Dh, Woh, bo, nullptr, out, X, beta,
                                    (int)D, (int)D, (int)D, (int)D, (int)D,
                                    0, 0, 0, 1.0f, 3, (int)GS);
}
